// Attention_987842478171
// MI455X (gfx1250) — compile-verified, matrix operand order checked by reading, not run on hardware
//
#include <hip/hip_runtime.h>

typedef _Float16 half_t;
typedef __attribute__((ext_vector_type(16))) _Float16 v16h;
typedef __attribute__((ext_vector_type(8)))  _Float16 v8h;
typedef __attribute__((ext_vector_type(4)))  _Float16 v4h;
typedef __attribute__((ext_vector_type(8)))  float    v8f;
typedef __attribute__((ext_vector_type(4)))  float    v4f;

#define BD 4
#define ND 2048
#define CD 512
#define HD 8
#define DD 64
#define QSCALE 0.125f   // D^-0.5 = 64^-0.5

// Per the CDNA5 16-bit A(16x32)/B(32x16) fragment layout, lane (m = lane&15,
// hi = lane>>4) holds two contiguous 8-half runs: K = hi*8+(0..7) and
// K = 16+hi*8+(0..7). So a "row-indexed" fragment is two b128 loads.
__device__ __forceinline__ v16h load_row_f16(const half_t* S, int row0, int ld,
                                             int k0, int lane) {
    const int m = lane & 15, hi = lane >> 4;
    const half_t* p = S + (size_t)(row0 + m) * ld + k0 + hi * 8;
    v8h lo = *(const v8h*)(p);        // K = hi*8 + 0..7
    v8h hh = *(const v8h*)(p + 16);   // K = 16 + hi*8 + 0..7
    return __builtin_shufflevector(lo, hh, 0, 1, 2, 3, 4, 5, 6, 7,
                                   8, 9, 10, 11, 12, 13, 14, 15);
}

__device__ __forceinline__ v8f wmma_f16(v16h a, v16h b, v8f c) {
    return __builtin_amdgcn_wmma_f32_16x16x32_f16(false, a, false, b, (short)0, c,
                                                  false, false);
}

// ---------------- Kernel 0: f32 -> f16 bulk convert (x, qkv_w, proj_w) ----------------
__global__ void __launch_bounds__(256) cvt_kernel(const float* __restrict__ src,
                                                  half_t* __restrict__ dst, int n4) {
    const int i = blockIdx.x * 256 + threadIdx.x;
    if (i < n4) {
        v4f v = ((const v4f*)src)[i];
        v4h h;
        h[0] = (half_t)v[0]; h[1] = (half_t)v[1];
        h[2] = (half_t)v[2]; h[3] = (half_t)v[3];
        ((v4h*)dst)[i] = h;
    }
}

// ---------------- Kernel 1: QKV projection ----------------
// One wave computes a 16x64 output strip (4 column tiles): one A fragment is
// reused by 4 WMMAs per K-step (10 b128 loads -> 4 WMMAs). The 64-wide group is
// 64-aligned, so it never straddles the q/k/v (512) or head (64) boundaries:
// t and h are wave-uniform. q,k -> [B,H,N,D] f16 (q pre-scaled); v -> [B,H,D,N].
__global__ void __launch_bounds__(256) qkv_kernel(const half_t* __restrict__ xh,
                                                  const half_t* __restrict__ wh,
                                                  const float* __restrict__ bias,
                                                  half_t* __restrict__ qf,
                                                  half_t* __restrict__ kf,
                                                  half_t* __restrict__ vt) {
    const int lane = threadIdx.x & 31;
    const int widx = threadIdx.x >> 5;
    const int wid  = blockIdx.x * 8 + widx;
    const int TO   = (3 * CD) / 64;           // 24 column groups of 64
    const int tm   = wid / TO;                // row tile over B*N/16 = 512
    const int og   = (wid % TO) * 64;         // first output column of the strip

    v8f acc[4] = {};
    for (int k0 = 0; k0 < CD; k0 += 32) {
        v16h a = load_row_f16(xh, tm * 16, CD, k0, lane);
#pragma unroll
        for (int c = 0; c < 4; ++c) {
            v16h b = load_row_f16(wh, og + c * 16, CD, k0, lane);  // B[k][n] = W[n][k]
            acc[c] = wmma_f16(a, b, acc[c]);
        }
    }

    const int hi = lane >> 4, ln = lane & 15;
    const int t = og / CD;                 // 0=q 1=k 2=v  (wave-uniform)
    const int h = (og % CD) / DD;          // head         (wave-uniform)
    if (t == 2) {
#pragma unroll
        for (int c = 0; c < 4; ++c) {
            const int d = c * 16 + ln;
            const float bv = bias[og + c * 16 + ln];
#pragma unroll
            for (int r = 0; r < 8; ++r) {
                const int row = tm * 16 + r + hi * 8;
                const int bb = row / ND, nn = row % ND;
                vt[(((size_t)bb * HD + h) * DD + d) * ND + nn] = (half_t)(acc[c][r] + bv);
            }
        }
    } else {
        half_t* dst = (t == 0) ? qf : kf;
        const float mul = (t == 0) ? QSCALE : 1.0f;
#pragma unroll
        for (int c = 0; c < 4; ++c) {
            const int d = c * 16 + ln;
            const float bv = bias[og + c * 16 + ln];
#pragma unroll
            for (int r = 0; r < 8; ++r) {
                const int row = tm * 16 + r + hi * 8;
                const int bb = row / ND, nn = row % ND;
                dst[(((size_t)bb * HD + h) * ND + nn) * DD + d] =
                    (half_t)((acc[c][r] + bv) * mul);
            }
        }
    }
}

// ---------------- Kernel 2: flash attention ----------------
// One wave = one 16-query tile of one (b,h). Streams 32-key blocks, online softmax.
__global__ void __launch_bounds__(256) attn_kernel(const half_t* __restrict__ qf,
                                                   const half_t* __restrict__ kf,
                                                   const half_t* __restrict__ vt,
                                                   half_t* __restrict__ of) {
    __shared__ __align__(16) half_t Plds[8][16][32];  // per-wave P staging
    const int lane = threadIdx.x & 31;
    const int widx = threadIdx.x >> 5;
    const int wid  = blockIdx.x * 8 + widx;
    const int QT   = ND / 16;            // 128 query tiles per (b,h)
    const int bh   = wid / QT;
    const int qt   = wid % QT;
    const int b    = bh / HD, h = bh % HD;
    const half_t* qb  = qf + (size_t)bh * ND * DD;
    const half_t* kb  = kf + (size_t)bh * ND * DD;
    const half_t* vtb = vt + (size_t)bh * DD * ND;   // [D, N] row-major
    const int hi = lane >> 4, ln = lane & 15;

    // q tile 16x64 (pre-scaled): two A fragments, kept in registers all along
    const v16h aq0 = load_row_f16(qb, qt * 16, DD, 0,  lane);
    const v16h aq1 = load_row_f16(qb, qt * 16, DD, 32, lane);

    float m_i[8], l_i[8];
    v8f acc[4] = {};                      // O tile 16x64 (4 column tiles), f32
#pragma unroll
    for (int r = 0; r < 8; ++r) { m_i[r] = -1e30f; l_i[r] = 0.0f; }

    for (int key0 = 0; key0 < ND; key0 += 32) {
        if (key0 + 32 < ND) {             // prefetch next key block (global_prefetch_b8)
            __builtin_prefetch(kb + (size_t)(key0 + 32) * DD, 0, 3);
            __builtin_prefetch(vtb + (size_t)(key0 + 32), 0, 3);
        }
        // S = q @ k^T for 32 keys: two 16x16 tiles, each 2 WMMAs over d
        v8f s0 = {}, s1 = {};
        {
            v16h bk;
            bk = load_row_f16(kb, key0,      DD, 0,  lane); s0 = wmma_f16(aq0, bk, s0);
            bk = load_row_f16(kb, key0,      DD, 32, lane); s0 = wmma_f16(aq1, bk, s0);
            bk = load_row_f16(kb, key0 + 16, DD, 0,  lane); s1 = wmma_f16(aq0, bk, s1);
            bk = load_row_f16(kb, key0 + 16, DD, 32, lane); s1 = wmma_f16(aq1, bk, s1);
        }
        // Online softmax: acc element r is row (r + 8*hi); columns stripe across
        // the 16 lanes of this half-wave -> butterfly reduce with masks 1,2,4,8.
#pragma unroll
        for (int r = 0; r < 8; ++r) {
            float mb = fmaxf(s0[r], s1[r]);
            mb = fmaxf(mb, __shfl_xor(mb, 1, 32));
            mb = fmaxf(mb, __shfl_xor(mb, 2, 32));
            mb = fmaxf(mb, __shfl_xor(mb, 4, 32));
            mb = fmaxf(mb, __shfl_xor(mb, 8, 32));
            const float mnew = fmaxf(m_i[r], mb);
            const float corr = __expf(m_i[r] - mnew);
            m_i[r] = mnew;
            const float p0 = __expf(s0[r] - mnew);
            const float p1 = __expf(s1[r] - mnew);
            float rs = p0 + p1;
            rs += __shfl_xor(rs, 1, 32);
            rs += __shfl_xor(rs, 2, 32);
            rs += __shfl_xor(rs, 4, 32);
            rs += __shfl_xor(rs, 8, 32);
            l_i[r] = l_i[r] * corr + rs;
            acc[0][r] *= corr; acc[1][r] *= corr;
            acc[2][r] *= corr; acc[3][r] *= corr;
            Plds[widx][r + hi * 8][ln]      = (half_t)p0;
            Plds[widx][r + hi * 8][ln + 16] = (half_t)p1;
        }
        asm volatile("s_wait_dscnt 0" ::: "memory");
        const v16h ap = load_row_f16(&Plds[widx][0][0], 0, 32, 0, lane);
        // O += P (16x32) @ V (32x64); V^T[d][key] row pattern -> two b128 loads/frag
#pragma unroll
        for (int t = 0; t < 4; ++t) {
            v16h bv = load_row_f16(vtb, t * 16, ND, key0, lane);
            acc[t] = wmma_f16(ap, bv, acc[t]);
        }
    }

    // Normalize and store in [B, N, H*D] order (ready for proj GEMM)
#pragma unroll
    for (int t = 0; t < 4; ++t) {
#pragma unroll
        for (int r = 0; r < 8; ++r) {
            const int row = qt * 16 + r + hi * 8;
            const int col = h * DD + t * 16 + ln;
            of[((size_t)b * ND + row) * CD + col] = (half_t)(acc[t][r] / l_i[r]);
        }
    }
}

// ---------------- Kernel 3: output projection ----------------
// Same 16x64-strip scheme as qkv_kernel; f32 output + bias straight to d_out.
__global__ void __launch_bounds__(256) proj_kernel(const half_t* __restrict__ a,
                                                   const half_t* __restrict__ wh,
                                                   const float* __restrict__ bias,
                                                   float* __restrict__ out) {
    const int lane = threadIdx.x & 31;
    const int widx = threadIdx.x >> 5;
    const int wid  = blockIdx.x * 8 + widx;
    const int TO   = CD / 64;            // 8 column groups of 64
    const int tm   = wid / TO;
    const int og   = (wid % TO) * 64;

    v8f acc[4] = {};
    for (int k0 = 0; k0 < CD; k0 += 32) {
        v16h af = load_row_f16(a, tm * 16, CD, k0, lane);
#pragma unroll
        for (int c = 0; c < 4; ++c) {
            v16h bf = load_row_f16(wh, og + c * 16, CD, k0, lane);  // B[k][n] = W[n][k]
            acc[c] = wmma_f16(af, bf, acc[c]);
        }
    }

    const int hi = lane >> 4, ln = lane & 15;
#pragma unroll
    for (int c = 0; c < 4; ++c) {
        const float bv = bias[og + c * 16 + ln];
#pragma unroll
        for (int r = 0; r < 8; ++r) {
            const int row = tm * 16 + r + hi * 8;
            out[(size_t)row * CD + og + c * 16 + ln] = acc[c][r] + bv;
        }
    }
}

extern "C" void kernel_launch(void* const* d_in, const int* in_sizes, int n_in,
                              void* d_out, int out_size, void* d_ws, size_t ws_size,
                              hipStream_t stream) {
    const float* x      = (const float*)d_in[0];
    const float* qkv_w  = (const float*)d_in[1];
    const float* qkv_b  = (const float*)d_in[2];
    const float* proj_w = (const float*)d_in[3];
    const float* proj_b = (const float*)d_in[4];
    float* out = (float*)d_out;

    const size_t xElems    = (size_t)BD * ND * CD;        // 4 Mi
    const size_t qkvwElems = (size_t)3 * CD * CD;         // 768 Ki
    const size_t pwElems   = (size_t)CD * CD;             // 256 Ki
    const size_t headElems = (size_t)BD * HD * ND * DD;   // 4 Mi

    half_t* xh  = (half_t*)d_ws;
    half_t* qwh = xh  + xElems;
    half_t* pwh = qwh + qkvwElems;
    half_t* qf  = pwh + pwElems;
    half_t* kf  = qf  + headElems;
    half_t* vt  = kf  + headElems;       // V transposed: [B,H,D,N]
    half_t* of  = vt  + headElems;       // attn out, f16 [B,N,C]

    // Kernel 0: one-time f32 -> f16 conversions (vectorized)
    cvt_kernel<<<(int)(xElems / 4 + 255) / 256, 256, 0, stream>>>(x, xh, (int)(xElems / 4));
    cvt_kernel<<<(int)(qkvwElems / 4 + 255) / 256, 256, 0, stream>>>(qkv_w, qwh,
                                                                     (int)(qkvwElems / 4));
    cvt_kernel<<<(int)(pwElems / 4 + 255) / 256, 256, 0, stream>>>(proj_w, pwh,
                                                                   (int)(pwElems / 4));

    // QKV: (B*N/16)*(3C/64) = 512*24 = 12288 strips, 8 waves/block -> 1536 blocks
    qkv_kernel<<<1536, 256, 0, stream>>>(xh, qwh, qkv_b, qf, kf, vt);
    // Attention: B*H*(N/16) = 4096 waves -> 512 blocks
    attn_kernel<<<512, 256, 0, stream>>>(qf, kf, vt, of);
    // Proj: (B*N/16)*(C/64) = 512*8 = 4096 strips -> 512 blocks
    proj_kernel<<<512, 256, 0, stream>>>(of, pwh, proj_b, out);
}
